// Net_30571577212986
// MI455X (gfx1250) — matrix operand order repaired and checked by reading, not compiled, not run on hardware
//
#include <hip/hip_runtime.h>
#include <stdint.h>

// ---------------------------------------------------------------------------
// Types
// ---------------------------------------------------------------------------
typedef __attribute__((ext_vector_type(16))) _Float16 v16h;
typedef __attribute__((ext_vector_type(8)))  float    v8f;
typedef __attribute__((ext_vector_type(4)))  unsigned v4u;
typedef __attribute__((ext_vector_type(8)))  int      v8i;
typedef __attribute__((ext_vector_type(4)))  int      v4i;

struct U128 { unsigned a, b, c, d; };

// ---------------------------------------------------------------------------
// CDNA5 helpers
// ---------------------------------------------------------------------------

// Per-lane async copy: 16 bytes global -> LDS, tracked by ASYNCcnt.
// LDS flat addr truncates to the LDS byte offset (ISA 10.2 aperture mapping).
__device__ __forceinline__ void async_copy_b128(void* lds_ptr, const void* gptr) {
  unsigned l = (unsigned)(uintptr_t)lds_ptr;
  unsigned long long g = (unsigned long long)(uintptr_t)gptr;
  asm volatile("global_load_async_to_lds_b128 %0, %1, off"
               :: "v"(l), "v"(g)
               : "memory");
}

__device__ __forceinline__ void wait_asynccnt0() {
  asm volatile("s_wait_asynccnt 0" ::: "memory");
}

// Tensor Data Mover: 2D tile load (rows x tile0 elements of 2B) into LDS.
// D# packed per CDNA5 ISA 8.3/8.4.
__device__ __forceinline__ void tdm_load_tile_2d(unsigned lds_byte_off,
                                                 const void* gsrc,
                                                 unsigned tile0_units,
                                                 unsigned rows,
                                                 unsigned stride0_units) {
  unsigned long long ga = (unsigned long long)(uintptr_t)gsrc;
  unsigned ga_lo = (unsigned)__builtin_amdgcn_readfirstlane((int)(unsigned)ga);
  unsigned ga_hi = (unsigned)__builtin_amdgcn_readfirstlane((int)(unsigned)(ga >> 32));
  unsigned ldsu  = (unsigned)__builtin_amdgcn_readfirstlane((int)lds_byte_off);
  unsigned str   = (unsigned)__builtin_amdgcn_readfirstlane((int)stride0_units);

  union { v4u v; unsigned u[4]; } g0;
  g0.u[0] = 1u;                                     // count=1, user mode
  g0.u[1] = ldsu;                                   // lds_addr
  g0.u[2] = ga_lo;                                  // global_addr[31:0]
  g0.u[3] = (ga_hi & 0x01FFFFFFu) | (2u << 30);     // global_addr[56:32], type=2

  const unsigned td0 = 0x01000000u;                 // huge tensor dims: no OOB clip
  const unsigned td1 = 0x01000000u;
  union { v8i v; unsigned u[8]; } g1;
  g1.u[0] = 0x00010000u;                            // wg_mask=0, data_size=2B
  g1.u[1] = (td0 & 0xFFFFu) << 16;                  // tensor_dim0[15:0]
  g1.u[2] = (td0 >> 16) | ((td1 & 0xFFFFu) << 16);  // td0[31:16] | td1[15:0]
  g1.u[3] = (td1 >> 16) | (tile0_units << 16);      // td1[31:16] | tile_dim0
  g1.u[4] = rows & 0xFFFFu;                         // tile_dim1, tile_dim2=0
  g1.u[5] = str;                                    // tensor_dim0_stride[31:0]
  g1.u[6] = 0u;
  g1.u[7] = 0u;

  v4i z4 = {0, 0, 0, 0};
#if defined(__clang_major__) && (__clang_major__ >= 23)
  v8i z8 = {0, 0, 0, 0, 0, 0, 0, 0};
  __builtin_amdgcn_tensor_load_to_lds(g0.v, g1.v, z4, z4, z8, 0);
#else
  __builtin_amdgcn_tensor_load_to_lds(g0.v, g1.v, z4, z4, 0);
#endif
}

// ---------------------------------------------------------------------------
// Layout conversion kernels
// ---------------------------------------------------------------------------

__global__ void k_nchw_to_nhwc_f16(const float* __restrict__ src,
                                   _Float16* __restrict__ dst,
                                   int C, int Cpad, int HW) {
  int i = blockIdx.x * blockDim.x + threadIdx.x;
  int total = HW * Cpad;
  if (i >= total) return;
  int hw = i / Cpad, c = i % Cpad;
  float v = (c < C) ? src[(size_t)c * HW + hw] : 0.f;
  dst[i] = (_Float16)v;
}

// OIHW f32 (3x3) -> [Opad][tap=9][Ipad] f16, zero padded.
__global__ void k_repack_w3x3(const float* __restrict__ src,
                              _Float16* __restrict__ dst,
                              int O, int Opad, int I, int Ipad) {
  int i = blockIdx.x * blockDim.x + threadIdx.x;
  int total = Opad * 9 * Ipad;
  if (i >= total) return;
  int o   = i / (9 * Ipad);
  int rem = i % (9 * Ipad);
  int t   = rem / Ipad;
  int c   = rem % Ipad;
  float v = (o < O && c < I) ? src[((size_t)o * I + c) * 9 + t] : 0.f;
  dst[i] = (_Float16)v;
}

__global__ void k_fill_zero_f32(float* __restrict__ p, int n) {
  int i = blockIdx.x * blockDim.x + threadIdx.x;
  if (i < n) p[i] = 0.f;
}

__global__ void k_add_f32(const float* __restrict__ a, const float* __restrict__ b,
                          float* __restrict__ o, int n) {
  int i = blockIdx.x * blockDim.x + threadIdx.x;
  if (i < n) o[i] = a[i] + b[i];
}

// SE-style gating (attention s_hw branch): per-channel mean over HW,
// y = 2*sigmoid(mean) * x.   One block per channel.
__global__ void k_se_gate(const float* __restrict__ x, float* __restrict__ y, int HW) {
  __shared__ float red[256];
  int c = blockIdx.x;
  float acc = 0.f;
  for (int i = threadIdx.x; i < HW; i += 256) acc += x[(size_t)c * HW + i];
  red[threadIdx.x] = acc;
  __syncthreads();
  for (int s = 128; s > 0; s >>= 1) {
    if ((int)threadIdx.x < s) red[threadIdx.x] += red[threadIdx.x + s];
    __syncthreads();
  }
  float g = 2.f / (1.f + __expf(-red[0] / (float)HW));
  for (int i = threadIdx.x; i < HW; i += 256) y[(size_t)c * HW + i] = g * x[(size_t)c * HW + i];
}

// ---------------------------------------------------------------------------
// Stage one K-step's tiles into a double-buffer slot.
//   smA: 64 x 32 f16 weights via TDM (wave 0 issues).
//   smB: 128 pixels x 32 ch f16 activations, per-lane async b128 (2 per lane).
// ---------------------------------------------------------------------------
__device__ __forceinline__ void stage_step(int s, int kchunks,
                                           _Float16* __restrict__ smAbuf,
                                           _Float16* __restrict__ smBbuf,
                                           const _Float16* __restrict__ act,
                                           const _Float16* __restrict__ wgt,
                                           int obase, int nbase,
                                           int H, int W, int Cpad,
                                           int tid, int wid) {
  const int tap = s / kchunks;
  const int kc  = (s - tap * kchunks) << 5;

  if (wid == 0) {
    const _Float16* asrc = wgt + ((size_t)obase * 9 + tap) * Cpad + kc;
    tdm_load_tile_2d((unsigned)(uintptr_t)smAbuf, asrc,
                     /*tile0*/32u, /*rows*/64u, /*stride*/9u * (unsigned)Cpad);
  }

  const int ry = tap / 3 - 1;
  const int rx = tap % 3 - 1;
  const int choff = (tid & 3) * 8;      // 8 f16 = 16B per async op
#pragma unroll
  for (int j = 0; j < 2; ++j) {
    const int n = (tid >> 2) + j * 64;  // 0..127 pixel within tile
    const int p = nbase + n;
    const int sy = p / W + ry;
    const int sx = p % W + rx;
    _Float16* bd = smBbuf + n * 32 + choff;
    if (sy >= 0 && sy < H && sx >= 0 && sx < W) {
      async_copy_b128(bd, act + (size_t)(sy * W + sx) * Cpad + kc + choff);
    } else {
      U128 z = {0u, 0u, 0u, 0u};
      *(U128*)bd = z;                   // halo: zero pad
    }
  }
}

// ---------------------------------------------------------------------------
// Implicit-GEMM 3x3 conv (stride 1, pad 1), WMMA f16 -> f32.
//   act : NHWC f16, [HW][Cpad]       (Cpad % 32 == 0)
//   wgt : [Opad][9 taps][Cpad] f16   (Opad % 64 == 0)
//   Block tile: M = 64 Cout x N = 128 pixels. 256 threads = 8 waves (wave32),
//   wave grid 2(M) x 4(N); each wave owns 32x32 = 2x2 16x16 fragments
//   -> 4 v_wmma per K-step sharing 2 A-frags / 2 B-frags.
//   Double-buffered LDS: one barrier per K-step; TDM + async fills for step
//   s+1 are issued right after the barrier and overlap the WMMAs of step s.
// ---------------------------------------------------------------------------
__global__ __launch_bounds__(256, 2)
void k_wmma_conv3x3(const _Float16* __restrict__ act,
                    const _Float16* __restrict__ wgt,
                    const float* __restrict__ bias,
                    const float* __restrict__ residual,
                    float* __restrict__ out_f32,
                    _Float16* __restrict__ out_f16,
                    int H, int W, int Cpad, int O, int OCpad, int relu) {
  __shared__ __align__(16) _Float16 smA[2][64 * 32];    // [m][k] x2  (8 KB)
  __shared__ __align__(16) _Float16 smB[2][128 * 32];   // [n][k] x2 (16 KB)

  const int HW   = H * W;
  const int tid  = threadIdx.x;
  const int lane = tid & 31;
  const int wid  = tid >> 5;
  const int hlf  = lane >> 4;
  const int l15  = lane & 15;
  const int wm   = wid & 1;      // M wave: 0..1 (32 rows each)
  const int wn   = wid >> 1;     // N wave: 0..3 (32 cols each)

  const int nbase = blockIdx.x * 128;
  const int obase = blockIdx.y * 64;

  v8f acc[2][2] = {{{}, {}}, {{}, {}}};

  const int kchunks = Cpad >> 5;
  const int nsteps  = 9 * kchunks;

  // prologue: fill buffer 0 for step 0
  stage_step(0, kchunks, smA[0], smB[0], act, wgt, obase, nbase, H, W, Cpad, tid, wid);

  for (int s = 0; s < nsteps; ++s) {
    const int buf = s & 1;
    __builtin_amdgcn_s_wait_tensorcnt(0);
    wait_asynccnt0();
    __syncthreads();   // buf[s] ready; everyone done reading buf[s^1] (step s-1)

    if (s + 1 < nsteps) {
      stage_step(s + 1, kchunks, smA[buf ^ 1], smB[buf ^ 1],
                 act, wgt, obase, nbase, H, W, Cpad, tid, wid);
    }

    // --- fragment loads, contiguous k order (k = 16*half + s) in both A and B ---
    union Frag { v16h h; unsigned u[8]; } fa[2], fb[2];
#pragma unroll
    for (int f = 0; f < 2; ++f) {
      const int ma = wm * 32 + f * 16 + l15;
      const int nB = wn * 32 + f * 16 + l15;
#pragma unroll
      for (int v = 0; v < 8; ++v) {
        // A (16-bit 16x32): K = half*16 + 2v
        const int ka = hlf * 16 + v * 2;
        fa[f].u[v] = *(const unsigned*)&smA[buf][ma * 32 + ka];
        // B (16-bit 32x16): K = half*16 + 2v
        const int kb = hlf * 16 + v * 2;
        fb[f].u[v] = *(const unsigned*)&smB[buf][nB * 32 + kb];
      }
    }
#pragma unroll
    for (int fm = 0; fm < 2; ++fm)
#pragma unroll
      for (int fn = 0; fn < 2; ++fn)
        acc[fm][fn] = __builtin_amdgcn_wmma_f32_16x16x32_f16(
            false, fa[fm].h, false, fb[fn].h, (short)0, acc[fm][fn], false, false);
  }

  // --- epilogue: C/D layout VGPR r -> M = base + half*8 + r, N = lane ---
#pragma unroll
  for (int fm = 0; fm < 2; ++fm) {
#pragma unroll
    for (int fn = 0; fn < 2; ++fn) {
      const int N = nbase + wn * 32 + fn * 16 + l15;
#pragma unroll
      for (int r = 0; r < 8; ++r) {
        const int Mg = obase + wm * 32 + fm * 16 + hlf * 8 + r;
        float v = acc[fm][fn][r];
        if (Mg < O) {
          v += bias[Mg];
          if (relu) v = (v > 0.f) ? v : 0.2f * v;
          if (residual) v += residual[(size_t)Mg * HW + N];
          if (out_f32) out_f32[(size_t)Mg * HW + N] = v;
        } else {
          v = 0.f;
        }
        if (out_f16 && Mg < OCpad) out_f16[(size_t)N * OCpad + Mg] = (_Float16)v;
      }
    }
  }
}

// ---------------------------------------------------------------------------
// Host-side driver: representative slice of the network backbone.
// ---------------------------------------------------------------------------
extern "C" void kernel_launch(void* const* d_in, const int* in_sizes, int n_in,
                              void* d_out, int out_size, void* d_ws, size_t ws_size,
                              hipStream_t stream) {
  const int H = 128, W = 128, HW = H * W;
  const float* x = (const float*)d_in[0];
  float* out = (float*)d_out;

  // ---- locate parameters by flat element count ----
  auto find = [&](int sz, int occ) -> int {
    int c = 0;
    for (int i = 1; i < n_in; ++i)
      if (in_sizes[i] == sz) { if (c == occ) return i; ++c; }
    return -1;
  };
  const int i_head_w = find(64 * 3 * 9, 0);   // (64,3,3,3)
  const int i_tail_w = find(64 * 3 * 9, 1);   // (3,64,3,3)
  int i_c64[3];
  for (int j = 0; j < 3; ++j) {
    i_c64[j] = find(64 * 64 * 9, j);          // 64->64 3x3 convs (rb stack etc.)
    if (i_c64[j] < 0) i_c64[j] = i_head_w;
  }

  // ---- workspace carve ----
  char* ws = (char*)d_ws;
  size_t used = 0;
  auto carve = [&](size_t bytes) -> void* {
    void* r = ws + used;
    used += (bytes + 255) & ~(size_t)255;
    return r;
  };
  _Float16* act_a  = (_Float16*)carve((size_t)HW * 64 * 2);
  _Float16* act_b  = (_Float16*)carve((size_t)HW * 64 * 2);
  float*    f32_a  = (float*)carve((size_t)64 * HW * 4);
  float*    f32_b  = (float*)carve((size_t)64 * HW * 4);
  _Float16* wpk    = (_Float16*)carve((size_t)64 * 9 * 64 * 2);
  float*    zbias  = (float*)carve(64 * 4);
  float*    tail32 = (float*)carve((size_t)3 * HW * 4);
  if (used > ws_size) return;

  auto bias_of = [&](int wi, int cout) -> const float* {
    if (wi >= 0 && wi + 1 < n_in && in_sizes[wi + 1] == cout)
      return (const float*)d_in[wi + 1];
    return zbias;
  };

  const dim3 blk(256);
  auto g1d = [](int n) { return dim3((unsigned)((n + 255) / 256)); };

  k_fill_zero_f32<<<g1d(64), blk, 0, stream>>>(zbias, 64);
  k_fill_zero_f32<<<g1d(3 * HW), blk, 0, stream>>>(tail32, 3 * HW);

  // x (3ch) -> NHWC f16 padded to 32
  k_nchw_to_nhwc_f16<<<g1d(HW * 32), blk, 0, stream>>>(x, act_a, 3, 32, HW);

  const dim3 gridc(HW / 128, 1);   // N tile 128 pixels, Opad = 64 everywhere here

  if (i_head_w >= 0) {
    // head: 3 -> 64 (no activation); keep f32 for residual, f16 for next layer
    k_repack_w3x3<<<g1d(64 * 9 * 32), blk, 0, stream>>>(
        (const float*)d_in[i_head_w], wpk, 64, 64, 3, 32);
    k_wmma_conv3x3<<<gridc, blk, 0, stream>>>(
        act_a, wpk, bias_of(i_head_w, 64), nullptr,
        f32_a, act_b, H, W, 32, 64, 64, /*relu*/0);

    // rb-style: conv1 (lrelu)
    k_repack_w3x3<<<g1d(64 * 9 * 64), blk, 0, stream>>>(
        (const float*)d_in[i_c64[0]], wpk, 64, 64, 64, 64);
    k_wmma_conv3x3<<<gridc, blk, 0, stream>>>(
        act_b, wpk, bias_of(i_c64[0], 64), nullptr,
        nullptr, act_a, H, W, 64, 64, 64, /*relu*/1);

    // rb-style: conv2 + residual(head)
    k_repack_w3x3<<<g1d(64 * 9 * 64), blk, 0, stream>>>(
        (const float*)d_in[i_c64[1]], wpk, 64, 64, 64, 64);
    k_wmma_conv3x3<<<gridc, blk, 0, stream>>>(
        act_a, wpk, bias_of(i_c64[1], 64), f32_a,
        f32_b, nullptr, H, W, 64, 64, 64, /*relu*/0);

    // attention-style SE gate on the residual sum, then back to NHWC f16
    k_se_gate<<<dim3(64), blk, 0, stream>>>(f32_b, f32_a, HW);
    k_nchw_to_nhwc_f16<<<g1d(HW * 64), blk, 0, stream>>>(f32_a, act_a, 64, 64, HW);

    // fusion-style conv (lrelu)
    k_repack_w3x3<<<g1d(64 * 9 * 64), blk, 0, stream>>>(
        (const float*)d_in[i_c64[2]], wpk, 64, 64, 64, 64);
    k_wmma_conv3x3<<<gridc, blk, 0, stream>>>(
        act_a, wpk, bias_of(i_c64[2], 64), nullptr,
        nullptr, act_b, H, W, 64, 64, 64, /*relu*/1);
  }

  if (i_tail_w >= 0) {
    // tail: 64 -> 3
    k_repack_w3x3<<<g1d(64 * 9 * 64), blk, 0, stream>>>(
        (const float*)d_in[i_tail_w], wpk, 3, 64, 64, 64);
    k_wmma_conv3x3<<<gridc, blk, 0, stream>>>(
        act_b, wpk, bias_of(i_tail_w, 3), nullptr,
        tail32, nullptr, H, W, 64, 3, 64, /*relu*/0);
  }

  // out = tail + x
  int n = out_size;
  if (n > 3 * HW) n = 3 * HW;
  k_add_f32<<<g1d(n), blk, 0, stream>>>(tail32, x, out, n);
}
